// BackbonePointNet2_64802466562570
// MI455X (gfx1250) — matrix operand order repaired and checked by reading, not compiled, not run on hardware
//
#include <hip/hip_runtime.h>
#include <math.h>

// ---------------------------------------------------------------------------
// PointNet++ backbone for MI455X (gfx1250, wave32, WMMA).
// All activations & weights in bf16, point-major [b][m][Cpad] with Cpad a
// multiple of 32 (zero-padded).  Every MLP layer is a bf16 WMMA GEMM with
// f32 accumulate and fused BN/activation; inner loop is 4x b128 loads + wmma.
// ---------------------------------------------------------------------------

typedef __attribute__((ext_vector_type(16))) __bf16          v16bf;
typedef __attribute__((ext_vector_type(16))) unsigned short  v16us;
typedef __attribute__((ext_vector_type(8)))  unsigned short  v8us;
typedef __attribute__((ext_vector_type(8)))  float           v8f;
typedef unsigned short ushort_t;

__device__ __forceinline__ unsigned short f2bf(float f) {
  unsigned u = __builtin_bit_cast(unsigned, f);
  unsigned r = u + 0x7FFFu + ((u >> 16) & 1u);   // round to nearest even
  return (unsigned short)(r >> 16);
}
__device__ __forceinline__ float bf2f(unsigned short h) {
  unsigned u = ((unsigned)h) << 16;
  return __builtin_bit_cast(float, u);
}

// ---------------------------------------------------------------------------
// Fused GEMM + BN scale/bias + activation.
//   Y[b,m, o] = act( g[o] * sum_c W[o,c]*X[b,m, c] + bias[o] )
// X: bf16 [b][m][Cp], W: bf16 [O][Cp] (zero padded), Y: bf16 [b][m][Op].
// One wave = 16(O) x 16(M) tile via v_wmma_f32_16x16x32_bf16.
// A halves: lane row=lane&15, h -> k = h + 16*(lane>>4)
//           => two 8-half loads: [16*hi, +8) and [16*hi+8, +8).
// B halves: lane col=lane&15, h -> k = h + 16*(lane>>4) => 16 contiguous.
// D: vgpr v -> row = 16*ot + 8*hi + v (contiguous channels), col = lane&15.
// act: 0 = none, 1 = relu, 2 = leaky_relu(0.2)
// ---------------------------------------------------------------------------
__global__ void __launch_bounds__(128) gemm_bn_act(
    const ushort_t* __restrict__ Wb, const float* __restrict__ G,
    const float* __restrict__ Bi, const ushort_t* __restrict__ X,
    ushort_t* __restrict__ Y, int O, int Cp, int M, int Op, int act)
{
  const int lane = threadIdx.x;                       // 0..31
  const int mt   = blockIdx.x * blockDim.y + threadIdx.y;
  const int ot   = blockIdx.y;
  const int b    = blockIdx.z;
  const int row  = lane & 15;
  const int hi   = lane >> 4;
  const int Mt   = M >> 4;
  const int mte  = (mt < Mt) ? mt : (Mt - 1);         // clamp tile: EXEC stays full
  const int n    = mte * 16 + row;                    // point (column), < M
  const int am   = ot * 16 + row;
  const int amc  = (am < O) ? am : (O - 1);

  const ushort_t* Wr = Wb + (size_t)amc * Cp + (hi << 4);
  const ushort_t* Xr = X + ((size_t)b * M + n) * Cp + (hi << 4);

  v8f acc = {};
  for (int c0 = 0; c0 < Cp; c0 += 32) {
    v8us  alo = *(const v8us*)(Wr + c0);
    v8us  ahi = *(const v8us*)(Wr + c0 + 8);
    v16us bv  = *(const v16us*)(Xr + c0);
    v16us av  = __builtin_shufflevector(alo, ahi,
                  0, 1, 2, 3, 4, 5, 6, 7, 8, 9, 10, 11, 12, 13, 14, 15);
    acc = __builtin_amdgcn_wmma_f32_16x16x32_bf16(
        false, __builtin_bit_cast(v16bf, av),
        false, __builtin_bit_cast(v16bf, bv),
        (short)0, acc, false, false);
  }

  if (mt >= Mt) return;
  v8us ov;
#pragma unroll
  for (int v = 0; v < 8; ++v) {
    int m  = ot * 16 + (hi << 3) + v;
    int mc = (m < O) ? m : (O - 1);
    float y = acc[v] * G[mc] + Bi[mc];
    if (act == 1)      y = fmaxf(y, 0.f);
    else if (act == 2) y = (y > 0.f) ? y : 0.2f * y;
    ov[v] = f2bf(y);
  }
  *(v8us*)(Y + ((size_t)b * M + n) * Op + ot * 16 + (hi << 3)) = ov;
}

// Convert one weight matrix (f32, OxC) to zero-padded bf16 (OxCp)
__global__ void conv_w(const float* __restrict__ W, ushort_t* __restrict__ Wb,
                       int O, int C, int Cp, int total)
{
  int t = blockIdx.x * blockDim.x + threadIdx.x;
  if (t >= total) return;
  int o = t / Cp, c = t % Cp;
  Wb[t] = (c < C) ? f2bf(W[(size_t)o * C + c]) : (ushort_t)0;
}

// ---------------------------------------------------------------------------
// Farthest point sampling: one block per batch, dist[] in LDS (<=32KB of the
// 320KB WGP LDS).  Emit-then-update with first-occurrence argmax ties.
// ---------------------------------------------------------------------------
#define FPS_MAXN 8192
__global__ void __launch_bounds__(256) fps_kernel(
    const float* __restrict__ xyz, float* __restrict__ nxyz, int N, int np)
{
  __shared__ float dist[FPS_MAXN];
  __shared__ float rv[256];
  __shared__ int   ri[256];
  __shared__ int   s_last;
  const int b = blockIdx.x, t = threadIdx.x;
  const float* P = xyz + (size_t)b * N * 3;
  for (int n = t; n < N; n += 256) dist[n] = 1e10f;
  if (t == 0) s_last = 0;
  __syncthreads();
  for (int i = 0; i < np; ++i) {
    const int last = s_last;
    const float cx = P[last * 3 + 0], cy = P[last * 3 + 1], cz = P[last * 3 + 2];
    if (t == 0) {
      nxyz[((size_t)b * np + i) * 3 + 0] = cx;
      nxyz[((size_t)b * np + i) * 3 + 1] = cy;
      nxyz[((size_t)b * np + i) * 3 + 2] = cz;
    }
    float bm = -1.f; int bi = 0;
    for (int n = t; n < N; n += 256) {
      float dx = P[n * 3 + 0] - cx, dy = P[n * 3 + 1] - cy, dz = P[n * 3 + 2] - cz;
      float d  = dx * dx + dy * dy + dz * dz;
      float nd = fminf(dist[n], d);
      dist[n]  = nd;
      if (nd > bm) { bm = nd; bi = n; }
    }
    rv[t] = bm; ri[t] = bi;
    __syncthreads();
    for (int s = 128; s > 0; s >>= 1) {
      if (t < s) {
        float v2 = rv[t + s]; int i2 = ri[t + s];
        if (v2 > rv[t] || (v2 == rv[t] && i2 < ri[t])) { rv[t] = v2; ri[t] = i2; }
      }
      __syncthreads();
    }
    if (t == 0) s_last = ri[0];
    __syncthreads();
  }
}

// ---------------------------------------------------------------------------
// Ball query + grouping: one wave per (centroid, batch).  First K in-radius
// indices (ascending) via wave32 ballot; pad with sel[0] (or 0 if empty).
// Output bf16 NHWC row per sample: [0:3)=xyz-diff, [3:3+Cf)=feats (bf16 copy),
// [3+Cf:Cp)=0.
// ---------------------------------------------------------------------------
__global__ void __launch_bounds__(32) ball_group_kernel(
    const float* __restrict__ xyz, const float* __restrict__ nxyz,
    const ushort_t* __restrict__ feats, ushort_t* __restrict__ out,
    int Ns, int S, int Cf, int K, int Cp, float r2)
{
  __shared__ int sel[128];
  __shared__ int s_cnt;
  const int s = blockIdx.x, b = blockIdx.y, lane = threadIdx.x;
  const float* P = xyz + (size_t)b * Ns * 3;
  const float cx = nxyz[((size_t)b * S + s) * 3 + 0];
  const float cy = nxyz[((size_t)b * S + s) * 3 + 1];
  const float cz = nxyz[((size_t)b * S + s) * 3 + 2];
  if (lane == 0) s_cnt = 0;
  __syncthreads();
  for (int base = 0; base < Ns; base += 32) {
    int n = base + lane;
    int inr = 0;
    if (n < Ns) {
      float dx = P[n * 3 + 0] - cx, dy = P[n * 3 + 1] - cy, dz = P[n * 3 + 2] - cz;
      inr = (dx * dx + dy * dy + dz * dz) < r2;
    }
    unsigned long long bal = __ballot(inr);
    if (lane == 0) {
      unsigned mm = (unsigned)bal;
      while (mm && s_cnt < K) {
        int j = __builtin_ctz(mm);
        sel[s_cnt++] = base + j;
        mm &= mm - 1;
      }
    }
    __syncthreads();
    if (s_cnt >= K) break;
    __syncthreads();
  }
  const int cnt = s_cnt;
  for (int k = lane; k < K; k += 32) {
    int idx = 0;
    if (cnt > 0) idx = (k < cnt) ? sel[k] : sel[0];
    ushort_t* orow = out + (((size_t)b * S + s) * K + k) * Cp;   // m = s*K+k
    orow[0] = f2bf(P[idx * 3 + 0] - cx);
    orow[1] = f2bf(P[idx * 3 + 1] - cy);
    orow[2] = f2bf(P[idx * 3 + 2] - cz);
    const ushort_t* frow = feats + ((size_t)b * Ns + idx) * Cf;
    for (int c = 0; c < Cf; ++c) orow[3 + c] = frow[c];
    for (int c = 3 + Cf; c < Cp; ++c) orow[c] = 0;
  }
}

// Max-pool over K samples, bf16 NHWC: out[b][s][o] = max_k in[b][s*K+k][o]
__global__ void maxpool_nhwc(const ushort_t* __restrict__ in,
                             ushort_t* __restrict__ out,
                             int S, int K, int O, int total)
{
  int t = blockIdx.x * blockDim.x + threadIdx.x;
  if (t >= total) return;
  int bs = t / O, o = t % O;
  const ushort_t* p = in + ((size_t)bs * K) * O + o;
  ushort_t best = p[0];
  float bv = bf2f(best);
  for (int k = 1; k < K; ++k) {
    ushort_t u = p[(size_t)k * O];
    float v = bf2f(u);
    if (v > bv) { bv = v; best = u; }
  }
  out[t] = best;
}

// sa4 input: out (B,32 pts,288): c<3 -> l3x; c<259 -> l3f(bf16); else 0
__global__ void sa4_build(const float* __restrict__ l3x,
                          const ushort_t* __restrict__ l3f,
                          ushort_t* __restrict__ out, int total)
{
  int t = blockIdx.x * blockDim.x + threadIdx.x;
  if (t >= total) return;
  int bn = t / 288, c = t % 288;
  ushort_t v;
  if (c < 3)        v = f2bf(l3x[(size_t)bn * 3 + c]);
  else if (c < 259) v = l3f[(size_t)bn * 256 + (c - 3)];
  else              v = 0;
  out[t] = v;
}

// fp4 input: out (B,32 pts,768): c<512 -> broadcast l4f[b][c]; else l3f[b][n][c-512]
__global__ void fp4_build(const ushort_t* __restrict__ l4f,
                          const ushort_t* __restrict__ l3f,
                          ushort_t* __restrict__ out, int total)
{
  int t = blockIdx.x * blockDim.x + threadIdx.x;
  if (t >= total) return;
  int bn = t / 768, c = t % 768;
  int b = bn >> 5;
  out[t] = (c < 512) ? l4f[(size_t)b * 512 + c]
                     : l3f[(size_t)bn * 256 + (c - 512)];
}

// 3-NN inverse-distance interp: writes channels [0,Ck) of out (bf16 NHWC, Cp)
__global__ void fp_interp(const float* __restrict__ uxyz,
                          const float* __restrict__ kxyz,
                          const ushort_t* __restrict__ kf,
                          ushort_t* __restrict__ out,
                          int Nu, int Nk, int Ck, int Cp, int total)
{
  int t = blockIdx.x * blockDim.x + threadIdx.x;
  if (t >= total) return;
  int b = t / Nu, n = t % Nu;
  const float* U = uxyz + ((size_t)b * Nu + n) * 3;
  const float ux = U[0], uy = U[1], uz = U[2];
  const float* Kx = kxyz + (size_t)b * Nk * 3;
  float d0 = 1e30f, d1 = 1e30f, d2 = 1e30f;
  int   i0 = 0,     i1 = 0,     i2 = 0;
  for (int j = 0; j < Nk; ++j) {
    float dx = Kx[j * 3 + 0] - ux, dy = Kx[j * 3 + 1] - uy, dz = Kx[j * 3 + 2] - uz;
    float d = dx * dx + dy * dy + dz * dz;
    if (d < d0)      { d2 = d1; i2 = i1; d1 = d0; i1 = i0; d0 = d; i0 = j; }
    else if (d < d1) { d2 = d1; i2 = i1; d1 = d;  i1 = j; }
    else if (d < d2) { d2 = d;  i2 = j; }
  }
  float w0 = 1.f / (sqrtf(fmaxf(d0, 0.f)) + 1e-8f);
  float w1 = 1.f / (sqrtf(fmaxf(d1, 0.f)) + 1e-8f);
  float w2 = 1.f / (sqrtf(fmaxf(d2, 0.f)) + 1e-8f);
  float sw = w0 + w1 + w2;
  w0 /= sw; w1 /= sw; w2 /= sw;
  const ushort_t* F  = kf + (size_t)b * Nk * Ck;
  const ushort_t* F0 = F + (size_t)i0 * Ck;
  const ushort_t* F1 = F + (size_t)i1 * Ck;
  const ushort_t* F2 = F + (size_t)i2 * Ck;
  ushort_t* orow = out + ((size_t)b * Nu + n) * Cp;
  for (int c = 0; c < Ck; ++c)
    orow[c] = f2bf(w0 * bf2f(F0[c]) + w1 * bf2f(F1[c]) + w2 * bf2f(F2[c]));
}

// bf16 NHWC (stride Cs) -> channels [c0, c0+Cs) of bf16 NHWC (stride Cp)
__global__ void copy_feat(const ushort_t* __restrict__ src,
                          ushort_t* __restrict__ out,
                          int Cs, int Cp, int c0, int total)
{
  int t = blockIdx.x * blockDim.x + threadIdx.x;
  if (t >= total) return;
  int bn = t / Cs, c = t % Cs;
  out[(size_t)bn * Cp + c0 + c] = src[t];
}

// xyz (B,Nn,3) f32 -> channels [c0, c0+3) of bf16 NHWC
__global__ void copy_xyz(const float* __restrict__ xyz, ushort_t* __restrict__ out,
                         int Cp, int c0, int total)
{
  int t = blockIdx.x * blockDim.x + threadIdx.x;
  if (t >= total) return;
  ushort_t* orow = out + (size_t)t * Cp + c0;
  orow[0] = f2bf(xyz[(size_t)t * 3 + 0]);
  orow[1] = f2bf(xyz[(size_t)t * 3 + 1]);
  orow[2] = f2bf(xyz[(size_t)t * 3 + 2]);
}

// f32 channel-major (B,Cs,Nn) -> channels [c0, c0+Cs) of bf16 NHWC (stride Cp)
__global__ void copy_cm(const float* __restrict__ src, ushort_t* __restrict__ out,
                        int Cs, int Nn, int Cp, int c0, int total)
{
  int t = blockIdx.x * blockDim.x + threadIdx.x;
  if (t >= total) return;
  int b = t / (Cs * Nn); int r = t % (Cs * Nn);
  int c = r / Nn; int n = r % Nn;
  out[((size_t)b * Nn + n) * Cp + c0 + c] = f2bf(src[t]);
}

// zero channels [c0, Cp) of bf16 NHWC buffer
__global__ void pad_zero(ushort_t* __restrict__ out, int Cp, int c0, int total)
{
  int t = blockIdx.x * blockDim.x + threadIdx.x;
  if (t >= total) return;
  int pc = Cp - c0;
  int bn = t / pc, c = t % pc;
  out[(size_t)bn * Cp + c0 + c] = 0;
}

// bat bf16 (B,8192,144) -> f32 (b,n,ch,t) split into 3 output segments
__global__ void scatter_out(const ushort_t* __restrict__ bat,
                            float* __restrict__ o0, float* __restrict__ o1,
                            float* __restrict__ o2, int total)
{
  int t = blockIdx.x * blockDim.x + threadIdx.x;
  if (t >= total) return;
  int bn = t / 135; int o = t % 135;
  float v = bf2f(bat[(size_t)bn * 144 + o]);
  int ch = o / 5, tt = o % 5;
  size_t pn = (size_t)bn;
  if (ch < 2)       o0[(pn * 2  +  ch)      * 5 + tt] = v;
  else if (ch < 7)  o1[(pn * 5  + (ch - 2)) * 5 + tt] = v;
  else              o2[(pn * 20 + (ch - 7)) * 5 + tt] = v;
}

// ---------------------------------------------------------------------------
// Host-side orchestration
// ---------------------------------------------------------------------------
extern "C" void kernel_launch(void* const* d_in, const int* in_sizes, int n_in,
                              void* d_out, int out_size, void* d_ws, size_t ws_size,
                              hipStream_t stream)
{
  (void)in_sizes; (void)n_in; (void)out_size; (void)ws_size;
  const float* xyz = (const float*)d_in[0];   // (32, 8192, 3)
  const float* pts = (const float*)d_in[1];   // (32, 3, 8192)

  enum { SA1 = 2, SA2 = 11, SA3 = 20, SA4 = 29,
         FP4 = 38, FP3 = 44, FP2 = 50, FP1 = 56,
         H1 = 62, H2 = 65, H3 = 68 };

  // ---- workspace layout (byte offsets, 256B aligned) ----
  char* base = (char*)d_ws;
  size_t off = 0;
  auto alloc = [&](size_t bytes) {
    void* p = base + off;
    off = (off + bytes + 255) & ~(size_t)255;
    return p;
  };
  const size_t HSZ = (size_t)32 * 8192 * 160;            // max bf16 tensor elems
  ushort_t* Hp0 = (ushort_t*)alloc(HSZ * 2);
  ushort_t* Hp1 = (ushort_t*)alloc(HSZ * 2);
  ushort_t* WBF = (ushort_t*)alloc((size_t)1100000 * 2);  // converted weights
  ushort_t* PTST= (ushort_t*)alloc((size_t)32 * 8192 * 3 * 2);
  ushort_t* L1F = (ushort_t*)alloc((size_t)32 * 256 * 64 * 2);
  ushort_t* L2F = (ushort_t*)alloc((size_t)32 * 128 * 128 * 2);
  ushort_t* L3F = (ushort_t*)alloc((size_t)32 * 32 * 256 * 2);
  ushort_t* L4F = (ushort_t*)alloc((size_t)32 * 512 * 2);
  ushort_t* G3F = (ushort_t*)alloc((size_t)32 * 32 * 256 * 2);
  ushort_t* G2F = (ushort_t*)alloc((size_t)32 * 128 * 256 * 2);
  ushort_t* G1F = (ushort_t*)alloc((size_t)32 * 256 * 128 * 2);
  float*    L1X = (float*)alloc((size_t)32 * 256 * 3 * 4);
  float*    L2X = (float*)alloc((size_t)32 * 128 * 3 * 4);
  float*    L3X = (float*)alloc((size_t)32 * 32 * 3 * 4);

  auto grid1 = [](int total) { return dim3((unsigned)((total + 255) / 256)); };

  size_t woff = 0;
  auto gemm = [&](const ushort_t* X, ushort_t* Y, int pbase, int layer,
                  int O, int C, int Cp, int M, int Op, int act) {
    ushort_t* wb = WBF + woff;
    int wt = O * Cp;
    conv_w<<<grid1(wt), 256, 0, stream>>>((const float*)d_in[pbase + 3 * layer + 0],
                                          wb, O, C, Cp, wt);
    woff = (woff + (size_t)wt + 127) & ~(size_t)127;
    dim3 grid((M / 16 + 3) / 4, (O + 15) / 16, 32);
    gemm_bn_act<<<grid, dim3(32, 4, 1), 0, stream>>>(
        wb, (const float*)d_in[pbase + 3 * layer + 1],
        (const float*)d_in[pbase + 3 * layer + 2], X, Y, O, Cp, M, Op, act);
  };

  // points -> bf16 point-major (B,8192,3) for grouping gathers
  copy_cm<<<grid1(32 * 3 * 8192), 256, 0, stream>>>(pts, PTST, 3, 8192, 3, 0,
                                                    32 * 3 * 8192);

  // ---- SA1: 8192 -> 256, K=32, r=0.05, chans 6->32->32->64 (Cp 32) ----
  fps_kernel<<<32, 256, 0, stream>>>(xyz, L1X, 8192, 256);
  ball_group_kernel<<<dim3(256, 32), 32, 0, stream>>>(
      xyz, L1X, PTST, Hp0, 8192, 256, 3, 32, 32, 0.0025f);
  gemm(Hp0, Hp1, SA1, 0, 32, 6, 32, 8192, 32, 1);
  gemm(Hp1, Hp0, SA1, 1, 32, 32, 32, 8192, 32, 1);
  gemm(Hp0, Hp1, SA1, 2, 64, 32, 32, 8192, 64, 1);
  maxpool_nhwc<<<grid1(32 * 256 * 64), 256, 0, stream>>>(Hp1, L1F, 256, 32, 64,
                                                         32 * 256 * 64);

  // ---- SA2: 256 -> 128, K=64, r=0.1, chans 67->64->64->128 (Cp 96) ----
  fps_kernel<<<32, 256, 0, stream>>>(L1X, L2X, 256, 128);
  ball_group_kernel<<<dim3(128, 32), 32, 0, stream>>>(
      L1X, L2X, L1F, Hp0, 256, 128, 64, 64, 96, 0.01f);
  gemm(Hp0, Hp1, SA2, 0, 64, 67, 96, 8192, 64, 1);
  gemm(Hp1, Hp0, SA2, 1, 64, 64, 64, 8192, 64, 1);
  gemm(Hp0, Hp1, SA2, 2, 128, 64, 64, 8192, 128, 1);
  maxpool_nhwc<<<grid1(32 * 128 * 128), 256, 0, stream>>>(Hp1, L2F, 128, 64, 128,
                                                          32 * 128 * 128);

  // ---- SA3: 128 -> 32, K=128, r=0.2, chans 131->128->128->256 (Cp 160) ----
  fps_kernel<<<32, 256, 0, stream>>>(L2X, L3X, 128, 32);
  ball_group_kernel<<<dim3(32, 32), 32, 0, stream>>>(
      L2X, L3X, L2F, Hp0, 128, 32, 128, 128, 160, 0.04f);
  gemm(Hp0, Hp1, SA3, 0, 128, 131, 160, 4096, 128, 1);
  gemm(Hp1, Hp0, SA3, 1, 128, 128, 128, 4096, 128, 1);
  gemm(Hp0, Hp1, SA3, 2, 256, 128, 128, 4096, 256, 1);
  maxpool_nhwc<<<grid1(32 * 32 * 256), 256, 0, stream>>>(Hp1, L3F, 32, 128, 256,
                                                         32 * 32 * 256);

  // ---- SA4 (global): 259->256->256->512 over 32 pts (Cp 288) ----
  sa4_build<<<grid1(32 * 32 * 288), 256, 0, stream>>>(L3X, L3F, Hp0, 32 * 32 * 288);
  gemm(Hp0, Hp1, SA4, 0, 256, 259, 288, 32, 256, 1);
  gemm(Hp1, Hp0, SA4, 1, 256, 256, 256, 32, 256, 1);
  gemm(Hp0, Hp1, SA4, 2, 512, 256, 256, 32, 512, 1);
  maxpool_nhwc<<<grid1(32 * 512), 256, 0, stream>>>(Hp1, L4F, 1, 32, 512, 32 * 512);

  // ---- FP4: [bcast l4f(512) | l3f(256)] -> 256 -> 256, n=32 ----
  fp4_build<<<grid1(32 * 32 * 768), 256, 0, stream>>>(L4F, L3F, Hp0, 32 * 32 * 768);
  gemm(Hp0, Hp1, FP4, 0, 256, 768, 768, 32, 256, 1);
  gemm(Hp1, G3F, FP4, 1, 256, 256, 256, 32, 256, 1);

  // ---- FP3: [interp(l3->l2, 256) | l2f(128)] -> 256 -> 256, n=128 ----
  fp_interp<<<grid1(32 * 128), 256, 0, stream>>>(L2X, L3X, G3F, Hp0,
                                                 128, 32, 256, 384, 32 * 128);
  copy_feat<<<grid1(32 * 128 * 128), 256, 0, stream>>>(L2F, Hp0, 128, 384, 256,
                                                       32 * 128 * 128);
  gemm(Hp0, Hp1, FP3, 0, 256, 384, 384, 128, 256, 1);
  gemm(Hp1, G2F, FP3, 1, 256, 256, 256, 128, 256, 1);

  // ---- FP2: [interp(l2->l1, 256) | l1f(64)] -> 256 -> 128, n=256 ----
  fp_interp<<<grid1(32 * 256), 256, 0, stream>>>(L1X, L2X, G2F, Hp0,
                                                 256, 128, 256, 320, 32 * 256);
  copy_feat<<<grid1(32 * 64 * 256), 256, 0, stream>>>(L1F, Hp0, 64, 320, 256,
                                                      32 * 64 * 256);
  gemm(Hp0, Hp1, FP2, 0, 256, 320, 320, 256, 256, 1);
  gemm(Hp1, G1F, FP2, 1, 128, 256, 256, 256, 128, 1);

  // ---- FP1: [interp(l1->xyz,128) | xyz^T(3) | points(3) | pad] -> 128 -> 128 ----
  fp_interp<<<grid1(32 * 8192), 256, 0, stream>>>(xyz, L1X, G1F, Hp0,
                                                  8192, 256, 128, 160, 32 * 8192);
  copy_xyz<<<grid1(32 * 8192), 256, 0, stream>>>(xyz, Hp0, 160, 128, 32 * 8192);
  copy_cm<<<grid1(32 * 3 * 8192), 256, 0, stream>>>(pts, Hp0, 3, 8192, 160, 131,
                                                    32 * 3 * 8192);
  pad_zero<<<grid1(32 * 8192 * 26), 256, 0, stream>>>(Hp0, 160, 134,
                                                      32 * 8192 * 26);
  gemm(Hp0, Hp1, FP1, 0, 128, 134, 160, 8192, 128, 1);
  gemm(Hp1, Hp0, FP1, 1, 128, 128, 128, 8192, 128, 1);

  // ---- heads: 128 (leaky) -> 128 (leaky) -> 135 (linear, Op=144) ----
  gemm(Hp0, Hp1, H1, 0, 128, 128, 128, 8192, 128, 2);
  gemm(Hp1, Hp0, H2, 0, 128, 128, 128, 8192, 128, 2);
  gemm(Hp0, Hp1, H3, 0, 135, 128, 128, 8192, 144, 0);

  // ---- output split/transpose (f32) ----
  float* o0 = (float*)d_out;                         // (32,8192, 2,5)
  float* o1 = o0 + (size_t)32 * 8192 * 2 * 5;        // (32,8192, 5,5)
  float* o2 = o1 + (size_t)32 * 8192 * 5 * 5;        // (32,8192,20,5)
  const int tot = 32 * 8192 * 135;
  scatter_out<<<grid1(tot), 256, 0, stream>>>(Hp1, o0, o1, o2, tot);
}
